// GaussianEncoder_35098472742933
// MI455X (gfx1250) — matrix operand order repaired and checked by reading, not compiled, not run on hardware
//
#include <hip/hip_runtime.h>

// ---------------------------------------------------------------------------
// CDNA5 (gfx1250) implementation: wave32 WMMA f32_16x16x32_f16 for all GEMMs,
// flash-attention with online softmax, fp32 VALU glue.
// Fast GEMM path uses b128 operand loads in a contiguous-K fragment order
// (k = 16*hi + s per lane, same for A and B) and 16x64 per-wave output strips.
// ---------------------------------------------------------------------------

typedef __attribute__((ext_vector_type(16))) _Float16 v16h;
typedef __attribute__((ext_vector_type(8)))  _Float16 v8h;
typedef __attribute__((ext_vector_type(8)))  float    v8f;
typedef _Float16 half_t;

#define DEV static __device__ __forceinline__

// A fragment (16x32 f16), unit k-stride: lane covers k run [16hi, 16hi+16)
// -> two b128 loads.
DEV v16h ld_afrag(const half_t* p, int hi) {
    v8h lo = *(const v8h*)(p + 16 * hi);
    v8h hp = *(const v8h*)(p + 16 * hi + 8);
    return __builtin_shufflevector(lo, hp, 0, 1, 2, 3, 4, 5, 6, 7,
                                   8, 9, 10, 11, 12, 13, 14, 15);
}
// B fragment (32x16 f16), unit k-stride along the pointed row:
// lane covers k run [16hi, 16hi+16) -> two b128 loads.
DEV v16h ld_bfrag(const half_t* p, int hi) {
    v8h lo = *(const v8h*)(p + 16 * hi);
    v8h hp = *(const v8h*)(p + 16 * hi + 8);
    return __builtin_shufflevector(lo, hp, 0, 1, 2, 3, 4, 5, 6, 7,
                                   8, 9, 10, 11, 12, 13, 14, 15);
}

// Block reduction (sum) over BS threads; reusable (trailing barrier).
template<int BS>
DEV float blk_sum(float v, float* red) {
    int t = threadIdx.x;
    red[t] = v; __syncthreads();
#pragma unroll
    for (int s = BS / 2; s > 0; s >>= 1) {
        if (t < s) red[t] += red[t + s];
        __syncthreads();
    }
    float r = red[0]; __syncthreads();
    return r;
}

// ---------------------------------------------------------------------------
// f32 -> f16 cast
// ---------------------------------------------------------------------------
__global__ void cast_f32_f16_kernel(const float* __restrict__ src,
                                    half_t* __restrict__ dst, int n) {
    int i = blockIdx.x * blockDim.x + threadIdx.x;
    if (i < n) dst[i] = (half_t)src[i];
}

// ---------------------------------------------------------------------------
// Embed: y = relu(LN(x @ W^T + b))  (H = 128, IN <= 64), one block per row.
// ---------------------------------------------------------------------------
__global__ void embed_ln_relu_kernel(const float* __restrict__ x, int IN,
                                     const float* __restrict__ W,
                                     const float* __restrict__ bias,
                                     const float* __restrict__ g,
                                     const float* __restrict__ beta,
                                     float* __restrict__ of,
                                     half_t* __restrict__ oh) {
    __shared__ float xs[64];
    __shared__ float red[128];
    int row = blockIdx.x, t = threadIdx.x;
    if (t < IN) xs[t] = x[row * IN + t];
    __syncthreads();
    float acc = bias[t];
    for (int i = 0; i < IN; ++i) acc += xs[i] * W[t * IN + i];
    float mean = blk_sum<128>(acc, red) * (1.0f / 128.0f);
    float d = acc - mean;
    float var = blk_sum<128>(d * d, red) * (1.0f / 128.0f);
    float y = d * rsqrtf(var + 1e-5f) * g[t] + beta[t];
    y = fmaxf(y, 0.0f);
    int idx = row * 128 + t;
    of[idx] = y; oh[idx] = (half_t)y;
}

// ---------------------------------------------------------------------------
// Residual + LN: y = LN(a + b), H = 128, one block per row.
// ---------------------------------------------------------------------------
__global__ void add_ln_kernel(const float* __restrict__ a,
                              const float* __restrict__ b,
                              const float* __restrict__ g,
                              const float* __restrict__ beta,
                              float* __restrict__ of,
                              half_t* __restrict__ oh) {
    __shared__ float red[128];
    int row = blockIdx.x, t = threadIdx.x;
    int idx = row * 128 + t;
    float v = a[idx] + b[idx];
    float mean = blk_sum<128>(v, red) * (1.0f / 128.0f);
    float d = v - mean;
    float var = blk_sum<128>(d * d, red) * (1.0f / 128.0f);
    float y = d * rsqrtf(var + 1e-5f) * g[t] + beta[t];
    of[idx] = y; oh[idx] = (half_t)y;
}

// ---------------------------------------------------------------------------
// FAST WMMA GEMM: C[m,n] = sum_k A[m*K+k] * B[n*K+k] + bias[n]
//   A: (M,K) f16 row-major; B: (N,K) f16 row-major (torch Linear W layout).
//   One wave per 16x64 output strip (A fragment reused over 4 WMMAs,
//   4 independent accumulators). K % 32 == 0, N % 64 == 0, M arbitrary.
//   Full-M tiles take a guard-free store path (wave-uniform branch).
// ---------------------------------------------------------------------------
template<bool OUTF, bool OUTH>
__global__ void gemm_wmma_fast_kernel(const half_t* __restrict__ A,
                                      const half_t* __restrict__ B,
                                      const float* __restrict__ bias,
                                      float* __restrict__ Cf,
                                      half_t* __restrict__ Ch,
                                      int M, int N, int K) {
    int lane = threadIdx.x & 31;
    int wid  = threadIdx.x >> 5;
    int nb4  = N >> 6;
    int tilesM = (M + 15) >> 4;
    int blk = blockIdx.x * (blockDim.x >> 5) + wid;
    if (blk >= tilesM * nb4) return;            // wave-uniform
    int m0 = (blk / nb4) << 4;
    int n0 = (blk % nb4) << 6;
    int hi = lane >> 4, l15 = lane & 15;
    int mrow = m0 + l15; if (mrow >= M) mrow = M - 1;
    const half_t* arow  = A + mrow * K;
    const half_t* brow0 = B + (n0 + l15) * K;
    const half_t* brow1 = B + (n0 + 16 + l15) * K;
    const half_t* brow2 = B + (n0 + 32 + l15) * K;
    const half_t* brow3 = B + (n0 + 48 + l15) * K;

    v8f acc[4] = {};
    for (int k0 = 0; k0 < K; k0 += 32) {
        __builtin_prefetch(brow0 + k0 + 64, 0, 0);
        __builtin_prefetch(brow1 + k0 + 64, 0, 0);
        __builtin_prefetch(brow2 + k0 + 64, 0, 0);
        __builtin_prefetch(brow3 + k0 + 64, 0, 0);
        v16h a  = ld_afrag(arow + k0, hi);
        v16h b0 = ld_bfrag(brow0 + k0, hi);
        v16h b1 = ld_bfrag(brow1 + k0, hi);
        v16h b2 = ld_bfrag(brow2 + k0, hi);
        v16h b3 = ld_bfrag(brow3 + k0, hi);
        acc[0] = __builtin_amdgcn_wmma_f32_16x16x32_f16(false, a, false, b0,
                                                        (short)0, acc[0], false, false);
        acc[1] = __builtin_amdgcn_wmma_f32_16x16x32_f16(false, a, false, b1,
                                                        (short)0, acc[1], false, false);
        acc[2] = __builtin_amdgcn_wmma_f32_16x16x32_f16(false, a, false, b2,
                                                        (short)0, acc[2], false, false);
        acc[3] = __builtin_amdgcn_wmma_f32_16x16x32_f16(false, a, false, b3,
                                                        (short)0, acc[3], false, false);
    }
    bool fullTile = (m0 + 16 <= M);             // wave-uniform
    if (fullTile) {
#pragma unroll
        for (int t = 0; t < 4; ++t) {
            int nc = n0 + 16 * t + l15;
            float bv = bias ? bias[nc] : 0.0f;
            int idx = (m0 + (hi ? 8 : 0)) * N + nc;
#pragma unroll
            for (int r = 0; r < 8; ++r) {
                float val = acc[t][r] + bv;
                if (OUTF) Cf[idx] = val;
                if (OUTH) Ch[idx] = (half_t)val;
                idx += N;
            }
        }
    } else {
#pragma unroll
        for (int t = 0; t < 4; ++t) {
            int nc = n0 + 16 * t + l15;
            float bv = bias ? bias[nc] : 0.0f;
#pragma unroll
            for (int r = 0; r < 8; ++r) {
                int mr = m0 + r + (hi ? 8 : 0);
                if (mr < M) {
                    float val = acc[t][r] + bv;
                    if (OUTF) Cf[mr * N + nc] = val;
                    if (OUTH) Ch[mr * N + nc] = (half_t)val;
                }
            }
        }
    }
}

// ---------------------------------------------------------------------------
// Generic strided batched WMMA GEMM — used only for the small batched NetVLAD
//   vlad[b] = a[b]^T @ xn[b]  matmul. A is stored transposed (unit k-stride)
// so the A operand uses the wide-load path; B stays scattered.
//   C[z,m,n] = sum_k A[z*bA + m*sAm + k] * B[z*bB + k*sBk + n*sBn]
// ---------------------------------------------------------------------------
__global__ void gemm_wmma_gen_kernel(const half_t* __restrict__ A,
                                     const half_t* __restrict__ B,
                                     float* __restrict__ Cf,
                                     int M, int N, int K,
                                     int sAm, int sBk, int sBn,
                                     int bA, int bB, int bC) {
    int lane = threadIdx.x & 31;
    int wid  = threadIdx.x >> 5;
    int tilesN = N >> 4;
    int tilesM = (M + 15) >> 4;
    int tile = blockIdx.x * (blockDim.x >> 5) + wid;
    if (tile >= tilesM * tilesN) return;
    int m0 = (tile / tilesN) << 4;
    int n0 = (tile % tilesN) << 4;
    const half_t* Ab = A + blockIdx.z * bA;
    const half_t* Bb = B + blockIdx.z * bB;
    int hi = lane >> 4, l15 = lane & 15;
    int mrow = m0 + l15; if (mrow >= M) mrow = M - 1;
    int ncol = n0 + l15;
    const half_t* arow = Ab + mrow * sAm;

    v8f acc = {};
    for (int k0 = 0; k0 < K; k0 += 32) {
        v16h a = ld_afrag(arow + k0, hi);
        v16h b;
#pragma unroll
        for (int v = 0; v < 8; ++v) {
            int kb = 2 * v + (hi ? 16 : 0) + k0;
            b[2 * v]     = Bb[kb * sBk + ncol * sBn];
            b[2 * v + 1] = Bb[(kb + 1) * sBk + ncol * sBn];
        }
        acc = __builtin_amdgcn_wmma_f32_16x16x32_f16(false, a, false, b,
                                                     (short)0, acc, false, false);
    }
    int cbase = blockIdx.z * bC;
    if (m0 + 16 <= M) {                          // wave-uniform
        int idx = cbase + (m0 + (hi ? 8 : 0)) * N + ncol;
#pragma unroll
        for (int r = 0; r < 8; ++r) { Cf[idx] = acc[r]; idx += N; }
    } else {
#pragma unroll
        for (int r = 0; r < 8; ++r) {
            int mr = m0 + r + (hi ? 8 : 0);
            if (mr < M) Cf[cbase + mr * N + ncol] = acc[r];
        }
    }
}

// ---------------------------------------------------------------------------
// Flash attention, one wave (32 threads) per (b, h, 16-query tile).
// Q/K/V: (B*Nseq, Hdim) f16, head h occupies columns [h*hd, h*hd+hd), hd=32.
// Q / K fragments: 2x b128 global loads each. V tile staged transposed in LDS
// (padded stride 34, conflict-free), P staged in LDS and re-read as 2x
// ds_load_b128. Online softmax with width-16 shuffles.
// ---------------------------------------------------------------------------
__global__ void flash_attn_kernel(const half_t* __restrict__ Q,
                                  const half_t* __restrict__ Kp,
                                  const half_t* __restrict__ Vp,
                                  half_t* __restrict__ O,
                                  int Nseq, int Hdim, int hd, float scale) {
    int lane = threadIdx.x;
    int hi   = lane >> 4;
    int l15  = lane & 15;
    int q0   = blockIdx.x * 16;
    int hoff = blockIdx.y * hd;
    int base = blockIdx.z * Nseq * Hdim;
    __shared__ __align__(16) half_t pT[16 * 32];     // P tile, row-major
    __shared__ half_t vT[32 * 34];                   // V tile transposed [d][key]

    // Q fragment (A layout 16x32): contiguous in d
    v16h aQ = ld_afrag(Q + base + (q0 + l15) * Hdim + hoff, hi);

    float m_i[8], l_i[8];
    v8f O0 = {}, O1 = {};
#pragma unroll
    for (int r = 0; r < 8; ++r) { m_i[r] = -1e30f; l_i[r] = 0.0f; }

    for (int j0 = 0; j0 < Nseq; j0 += 32) {
        // K fragments (B layout): contiguous in d
        v16h bK0 = ld_bfrag(Kp + base + (j0 + l15) * Hdim + hoff, hi);
        v16h bK1 = ld_bfrag(Kp + base + (j0 + 16 + l15) * Hdim + hoff, hi);
        // Cooperative V load: lane <-> key row, 4x b128 global loads
        half_t vals[32];
        {
            const v8h* vr = (const v8h*)(Vp + base + (j0 + lane) * Hdim + hoff);
            v8h r0 = vr[0], r1 = vr[1], r2 = vr[2], r3 = vr[3];
#pragma unroll
            for (int i = 0; i < 8; ++i) {
                vals[i] = r0[i]; vals[8 + i] = r1[i];
                vals[16 + i] = r2[i]; vals[24 + i] = r3[i];
            }
        }
        v8f S0 = {}, S1 = {};
        S0 = __builtin_amdgcn_wmma_f32_16x16x32_f16(false, aQ, false, bK0,
                                                    (short)0, S0, false, false);
        S1 = __builtin_amdgcn_wmma_f32_16x16x32_f16(false, aQ, false, bK1,
                                                    (short)0, S1, false, false);
        // online softmax per row (row = r + 8*hi lives in one 16-lane half)
        float p0[8], p1[8];
#pragma unroll
        for (int r = 0; r < 8; ++r) {
            float s0 = scale * S0[r], s1 = scale * S1[r];
            float mx = fmaxf(s0, s1);
#pragma unroll
            for (int off = 1; off < 16; off <<= 1)
                mx = fmaxf(mx, __shfl_xor(mx, off, 16));
            float mn = fmaxf(m_i[r], mx);
            float e0 = __expf(s0 - mn), e1 = __expf(s1 - mn);
            float rs = e0 + e1;
#pragma unroll
            for (int off = 1; off < 16; off <<= 1)
                rs += __shfl_xor(rs, off, 16);
            float alpha = __expf(m_i[r] - mn);
            l_i[r] = l_i[r] * alpha + rs;
            m_i[r] = mn;
            O0[r] *= alpha; O1[r] *= alpha;
            p0[r] = e0; p1[r] = e1;
        }
        // stage P (C layout -> row-major 16x32) and V (transposed, padded)
#pragma unroll
        for (int r = 0; r < 8; ++r) {
            int row = r + (hi ? 8 : 0);
            pT[row * 32 + l15]      = (half_t)p0[r];
            pT[row * 32 + 16 + l15] = (half_t)p1[r];
        }
#pragma unroll
        for (int d = 0; d < 32; ++d) vT[d * 34 + lane] = vals[d];
        __syncthreads();
        // P as A fragment: 2x ds_load_b128
        v16h aP = ld_afrag(pT + l15 * 32, hi);
        // V fragments from transposed tile: adjacent-k pairs are adjacent LDS
        v16h bV0, bV1;
        {
            const half_t* vrow0 = &vT[l15 * 34];
            const half_t* vrow1 = &vT[(16 + l15) * 34];
#pragma unroll
            for (int v = 0; v < 8; ++v) {
                int kb = 2 * v + (hi ? 16 : 0);
                bV0[2 * v]     = vrow0[kb];
                bV0[2 * v + 1] = vrow0[kb + 1];
                bV1[2 * v]     = vrow1[kb];
                bV1[2 * v + 1] = vrow1[kb + 1];
            }
        }
        O0 = __builtin_amdgcn_wmma_f32_16x16x32_f16(false, aP, false, bV0,
                                                    (short)0, O0, false, false);
        O1 = __builtin_amdgcn_wmma_f32_16x16x32_f16(false, aP, false, bV1,
                                                    (short)0, O1, false, false);
        __syncthreads();
    }
    // normalize + store (C layout)
#pragma unroll
    for (int r = 0; r < 8; ++r) {
        int row = q0 + r + (hi ? 8 : 0);
        float inv = 1.0f / l_i[r];
        int o = base + row * Hdim + hoff;
        O[o + l15]      = (half_t)(O0[r] * inv);
        O[o + 16 + l15] = (half_t)(O1[r] * inv);
    }
}

// ---------------------------------------------------------------------------
// Misc glue kernels
// ---------------------------------------------------------------------------
__global__ void colmean_kernel(const float* __restrict__ x, float* __restrict__ dst,
                               int Nseq, int off) {
    int b = blockIdx.x, t = threadIdx.x;           // 128 threads
    const float* p = x + b * Nseq * 128 + t;
    float acc = 0.0f;
    for (int n = 0; n < Nseq; ++n) acc += p[n * 128];
    dst[b * 256 + off + t] = acc * (1.0f / Nseq);
}

__global__ void router_r1_kernel(const float* __restrict__ ri,
                                 const float* __restrict__ W,
                                 const float* __restrict__ bias,
                                 const float* __restrict__ g,
                                 const float* __restrict__ beta,
                                 float* __restrict__ out) {
    __shared__ float red[128];
    int b = blockIdx.x, t = threadIdx.x;
    const float* x = ri + b * 256;
    float acc = bias[t];
    for (int i = 0; i < 256; ++i) acc += x[i] * W[t * 256 + i];
    float mean = blk_sum<128>(acc, red) * (1.0f / 128.0f);
    float d = acc - mean;
    float var = blk_sum<128>(d * d, red) * (1.0f / 128.0f);
    float y = d * rsqrtf(var + 1e-5f) * g[t] + beta[t];
    out[b * 128 + t] = fmaxf(y, 0.0f);
}

__global__ void router_r2_kernel(const float* __restrict__ r1o,
                                 const float* __restrict__ W,
                                 const float* __restrict__ bias,
                                 float* __restrict__ rw_out) {
    __shared__ float sv[3];
    int b = blockIdx.x, t = threadIdx.x;           // 32 threads
    if (t < 3) {
        const float* x = r1o + b * 128;
        float acc = bias[t];
        for (int i = 0; i < 128; ++i) acc += x[i] * W[t * 128 + i];
        sv[t] = acc;
    }
    __syncthreads();
    if (t == 0) {
        float mx = fmaxf(sv[0], fmaxf(sv[1], sv[2]));
        float e0 = __expf(sv[0] - mx), e1 = __expf(sv[1] - mx), e2 = __expf(sv[2] - mx);
        float inv = 1.0f / (e0 + e1 + e2);
        rw_out[b * 3 + 0] = e0 * inv;
        rw_out[b * 3 + 1] = e1 * inv;
        rw_out[b * 3 + 2] = e2 * inv;
    }
}

__global__ void concat_h_kernel(const half_t* __restrict__ a,
                                const half_t* __restrict__ b,
                                half_t* __restrict__ dst, int M) {
    int i = blockIdx.x * blockDim.x + threadIdx.x;
    if (i >= M * 256) return;
    int m = i >> 8; int c = i & 255;
    dst[i] = (c < 128) ? a[m * 128 + c] : b[m * 128 + (c - 128)];
}

__global__ void gate_fuse_kernel(const float* __restrict__ f1,
                                 const float* __restrict__ f2,
                                 const float* __restrict__ glin,
                                 float* __restrict__ fused, int n) {
    int i = blockIdx.x * blockDim.x + threadIdx.x;
    if (i >= n) return;
    float g = 1.0f / (1.0f + __expf(-glin[i]));
    fused[i] = f1[i] * g + f2[i] * (1.0f - g);
}

__global__ void l2norm_rows_kernel(const float* __restrict__ x,
                                   float* __restrict__ of,
                                   half_t* __restrict__ oh) {
    __shared__ float red[128];
    int row = blockIdx.x, t = threadIdx.x;
    int idx = row * 128 + t;
    float v = x[idx];
    float ss = blk_sum<128>(v * v, red);
    float y = v * (1.0f / fmaxf(sqrtf(ss), 1e-12f));
    of[idx] = y; oh[idx] = (half_t)y;
}

// Softmax over K=64 clusters; writes row-major f32 (for the asum reduction)
// and TRANSPOSED f16 (B*64, Nseq) so the VLAD GEMM's A operand is unit-stride.
__global__ void softmax64_kernel(const float* __restrict__ logits,
                                 float* __restrict__ af,
                                 half_t* __restrict__ ah_t) {
    __shared__ float red[64];
    int row = blockIdx.x, t = threadIdx.x;
    int idx = row * 64 + t;
    float v = logits[idx];
    red[t] = v; __syncthreads();
    for (int s = 32; s > 0; s >>= 1) { if (t < s) red[t] = fmaxf(red[t], red[t + s]); __syncthreads(); }
    float mx = red[0]; __syncthreads();
    float e = __expf(v - mx);
    red[t] = e; __syncthreads();
    for (int s = 32; s > 0; s >>= 1) { if (t < s) red[t] += red[t + s]; __syncthreads(); }
    float a = e / red[0];
    af[idx] = a;
    int b = row >> 11, n = row & 2047;            // Nseq = 2048
    ah_t[((b << 6) + t) * 2048 + n] = (half_t)a;
}

__global__ void asum_kernel(const float* __restrict__ af, float* __restrict__ asum,
                            int Nseq) {
    int b = blockIdx.x, k = threadIdx.x;           // 64 threads
    const float* p = af + b * Nseq * 64 + k;
    float acc = 0.0f;
    for (int n = 0; n < Nseq; ++n) acc += p[n * 64];
    asum[b * 64 + k] = acc;
}

__global__ void vlad_post1_kernel(float* __restrict__ vlad,
                                  const float* __restrict__ asum,
                                  const float* __restrict__ cent) {
    __shared__ float red[128];
    int b = blockIdx.x >> 6, k = blockIdx.x & 63, d = threadIdx.x;
    int idx = b * 8192 + k * 128 + d;
    float v = vlad[idx] - asum[b * 64 + k] * cent[k * 128 + d];
    float ss = blk_sum<128>(v * v, red);
    vlad[idx] = v * (1.0f / fmaxf(sqrtf(ss), 1e-12f));
}

__global__ void vlad_post2_kernel(const float* __restrict__ vlad,
                                  half_t* __restrict__ vh) {
    __shared__ float red[256];
    int b = blockIdx.x, t = threadIdx.x;
    const float* p = vlad + b * 8192;
    float ss = 0.0f;
    for (int i = t; i < 8192; i += 256) { float v = p[i]; ss += v * v; }
    ss = blk_sum<256>(ss, red);
    float inv = 1.0f / fmaxf(sqrtf(ss), 1e-12f);
    for (int i = t; i < 8192; i += 256) vh[b * 8192 + i] = (half_t)(p[i] * inv);
}

__global__ void desc_finish_kernel(const float* __restrict__ lin,
                                   const float* __restrict__ g,
                                   const float* __restrict__ beta,
                                   float* __restrict__ out) {
    __shared__ float red[256];
    int b = blockIdx.x, t = threadIdx.x;
    float v = lin[b * 256 + t];
    float mean = blk_sum<256>(v, red) * (1.0f / 256.0f);
    float d = v - mean;
    float var = blk_sum<256>(d * d, red) * (1.0f / 256.0f);
    float y = d * rsqrtf(var + 1e-5f) * g[t] + beta[t];
    float ss = blk_sum<256>(y * y, red);
    out[b * 256 + t] = y * (1.0f / fmaxf(sqrtf(ss), 1e-12f));
}

__global__ void final_kernel(float* __restrict__ d_out) {
    __shared__ float red[256];
    int b = blockIdx.x, t = threadIdx.x;
    const float* rw = d_out + 2048;
    float v = rw[b * 3 + 0] * d_out[2072 + b * 256 + t] +
              rw[b * 3 + 1] * d_out[4120 + b * 256 + t] +
              rw[b * 3 + 2] * d_out[6168 + b * 256 + t];
    float ss = blk_sum<256>(v * v, red);
    d_out[b * 256 + t] = v * (1.0f / fmaxf(sqrtf(ss), 1e-12f));
}

// ---------------------------------------------------------------------------
// Host orchestration
// ---------------------------------------------------------------------------
extern "C" void kernel_launch(void* const* d_in, const int* in_sizes, int n_in,
                              void* d_out, int out_size, void* d_ws, size_t ws_size,
                              hipStream_t stream) {
    (void)in_sizes; (void)n_in; (void)out_size; (void)ws_size;
    const int B = 8, Nseq = 2048, H = 128, HEADS = 4, HD = 32, K = 64, OUT = 256;
    const int M = B * Nseq;                // 16384
    const long MH = (long)M * H;           // 2097152

    const float* gaussians = (const float*)d_in[0];
    const float* visual    = (const float*)d_in[1];
    const float* pg_w = (const float*)d_in[2],  *pg_b = (const float*)d_in[3];
    const float* pg_g = (const float*)d_in[4],  *pg_be = (const float*)d_in[5];
    const float* pv_w = (const float*)d_in[6],  *pv_b = (const float*)d_in[7];
    const float* pv_g = (const float*)d_in[8],  *pv_be = (const float*)d_in[9];
    const float* g2v_in_w = (const float*)d_in[10], *g2v_in_b = (const float*)d_in[11];
    const float* g2v_out_w = (const float*)d_in[12], *g2v_out_b = (const float*)d_in[13];
    const float* v2g_in_w = (const float*)d_in[14], *v2g_in_b = (const float*)d_in[15];
    const float* v2g_out_w = (const float*)d_in[16], *v2g_out_b = (const float*)d_in[17];
    const float* n1_g = (const float*)d_in[18], *n1_b = (const float*)d_in[19];
    const float* n2_g = (const float*)d_in[20], *n2_b = (const float*)d_in[21];
    const float* gate_w = (const float*)d_in[22], *gate_b = (const float*)d_in[23];
    const float* r1_w = (const float*)d_in[24], *r1_b = (const float*)d_in[25];
    const float* r1_g = (const float*)d_in[26], *r1_be = (const float*)d_in[27];
    const float* r2_w = (const float*)d_in[28], *r2_b = (const float*)d_in[29];
    // branch params start at 30 (geo), 37 (vis), 44 (fus):
    //   +0 cent, +1 conv_w, +2 conv_b, +3 bn_w, +4 bn_b, +5 bn_g, +6 bn_beta

    char* wsp = (char*)d_ws;
    auto alloc = [&](size_t bytes) -> void* {
        void* p = (void*)wsp;
        wsp += (bytes + 255) & ~(size_t)255;
        return p;
    };
    float*  g_emb   = (float*)alloc(MH * 4);
    float*  v_emb   = (float*)alloc(MH * 4);
    float*  mha_lin = (float*)alloc(MH * 4);
    float*  f1      = (float*)alloc(MH * 4);
    float*  f2      = (float*)alloc(MH * 4);
    float*  gatelin = (float*)alloc(MH * 4);
    float*  fused   = (float*)alloc(MH * 4);
    float*  xn      = (float*)alloc(MH * 4);
    float*  logits  = (float*)alloc((long)M * K * 4);
    float*  a_f     = (float*)alloc((long)M * K * 4);
    float*  ri      = (float*)alloc(B * 256 * 4);
    float*  r1o     = (float*)alloc(B * 128 * 4);
    float*  asum    = (float*)alloc(B * 64 * 4);
    float*  vlad    = (float*)alloc((long)B * 8192 * 4);
    float*  bnlin   = (float*)alloc(B * 256 * 4);
    half_t* g_emb_h = (half_t*)alloc(MH * 2);
    half_t* v_emb_h = (half_t*)alloc(MH * 2);
    half_t* q_h     = (half_t*)alloc(MH * 2);
    half_t* k_h     = (half_t*)alloc(MH * 2);
    half_t* v_h     = (half_t*)alloc(MH * 2);
    half_t* attn_h  = (half_t*)alloc(MH * 2);
    half_t* f1_h    = (half_t*)alloc(MH * 2);
    half_t* f2_h    = (half_t*)alloc(MH * 2);
    half_t* cat_h   = (half_t*)alloc(MH * 2 * 2);
    half_t* xn_h    = (half_t*)alloc(MH * 2);
    half_t* a_ht    = (half_t*)alloc((long)M * K * 2);   // transposed (B*64, Nseq)
    half_t* vlad_h  = (half_t*)alloc((long)B * 8192 * 2);
    half_t* w_in_h  = (half_t*)alloc(384 * 128 * 2);
    half_t* w_out_h = (half_t*)alloc(128 * 128 * 2);
    half_t* gate_w_h = (half_t*)alloc(128 * 256 * 2);
    half_t* conv_w_h = (half_t*)alloc(64 * 128 * 2);
    half_t* bn_w_h  = (half_t*)alloc((long)OUT * 8192 * 2);

    auto castw = [&](const float* s, half_t* dst, int n) {
        cast_f32_f16_kernel<<<(n + 255) / 256, 256, 0, stream>>>(s, dst, n);
    };
    // Fast GEMM launchers (unit k-stride, N % 64 == 0)
    auto gemmF32 = [&](const half_t* A, const half_t* Bm, const float* bias,
                       float* Cf, int Mm, int Nn, int Kk) {
        int blks = ((Mm + 15) / 16) * (Nn / 64);
        gemm_wmma_fast_kernel<true, false><<<(blks + 7) / 8, 256, 0, stream>>>(
            A, Bm, bias, Cf, nullptr, Mm, Nn, Kk);
    };
    auto gemmF16 = [&](const half_t* A, const half_t* Bm, const float* bias,
                       half_t* Ch, int Mm, int Nn, int Kk) {
        int blks = ((Mm + 15) / 16) * (Nn / 64);
        gemm_wmma_fast_kernel<false, true><<<(blks + 7) / 8, 256, 0, stream>>>(
            A, Bm, bias, nullptr, Ch, Mm, Nn, Kk);
    };

    // -------- embeddings --------
    embed_ln_relu_kernel<<<M, 128, 0, stream>>>(gaussians, 14, pg_w, pg_b, pg_g, pg_be,
                                                g_emb, g_emb_h);
    embed_ln_relu_kernel<<<M, 128, 0, stream>>>(visual, 64, pv_w, pv_b, pv_g, pv_be,
                                                v_emb, v_emb_h);

    // -------- router --------
    colmean_kernel<<<B, 128, 0, stream>>>(g_emb, ri, Nseq, 0);
    colmean_kernel<<<B, 128, 0, stream>>>(v_emb, ri, Nseq, 128);
    router_r1_kernel<<<B, 128, 0, stream>>>(ri, r1_w, r1_b, r1_g, r1_be, r1o);
    router_r2_kernel<<<B, 32, 0, stream>>>(r1o, r2_w, r2_b, (float*)d_out + 2048);

    // -------- cross attention (both directions) --------
    const float scale = 0.17677669529663687f;   // 1/sqrt(32)
    auto mha = [&](const half_t* qsrc, const half_t* kvsrc,
                   const float* in_w, const float* in_b,
                   const float* out_w, const float* out_b,
                   const float* res, const float* ln_g, const float* ln_b,
                   float* fo, half_t* fo_h) {
        castw(in_w, w_in_h, 384 * 128);
        castw(out_w, w_out_h, 128 * 128);
        gemmF16(qsrc, w_in_h, in_b, q_h, M, 128, 128);
        gemmF16(kvsrc, w_in_h + 128 * 128, in_b + 128, k_h, M, 128, 128);
        gemmF16(kvsrc, w_in_h + 256 * 128, in_b + 256, v_h, M, 128, 128);
        flash_attn_kernel<<<dim3(Nseq / 16, HEADS, B), 32, 0, stream>>>(
            q_h, k_h, v_h, attn_h, Nseq, H, HD, scale);
        gemmF32(attn_h, w_out_h, out_b, mha_lin, M, 128, 128);
        add_ln_kernel<<<M, 128, 0, stream>>>(res, mha_lin, ln_g, ln_b, fo, fo_h);
    };
    mha(g_emb_h, v_emb_h, g2v_in_w, g2v_in_b, g2v_out_w, g2v_out_b,
        g_emb, n1_g, n1_b, f1, f1_h);
    mha(v_emb_h, g_emb_h, v2g_in_w, v2g_in_b, v2g_out_w, v2g_out_b,
        v_emb, n2_g, n2_b, f2, f2_h);

    // -------- gated fusion --------
    castw(gate_w, gate_w_h, 128 * 256);
    concat_h_kernel<<<(int)((MH * 2 + 255) / 256), 256, 0, stream>>>(f1_h, f2_h, cat_h, M);
    gemmF32(cat_h, gate_w_h, gate_b, gatelin, M, 128, 256);
    gate_fuse_kernel<<<(int)((MH + 255) / 256), 256, 0, stream>>>(f1, f2, gatelin,
                                                                  fused, (int)MH);

    // -------- NetVLAD branches --------
    auto branch = [&](const float* x, int pbase, int slot) {
        const float* cent    = (const float*)d_in[pbase + 0];
        const float* conv_w  = (const float*)d_in[pbase + 1];
        const float* conv_b  = (const float*)d_in[pbase + 2];
        const float* bn_w    = (const float*)d_in[pbase + 3];
        const float* bn_b    = (const float*)d_in[pbase + 4];
        const float* bn_g    = (const float*)d_in[pbase + 5];
        const float* bn_be   = (const float*)d_in[pbase + 6];
        l2norm_rows_kernel<<<M, 128, 0, stream>>>(x, xn, xn_h);
        castw(conv_w, conv_w_h, 64 * 128);
        gemmF32(xn_h, conv_w_h, conv_b, logits, M, 64, 128);
        softmax64_kernel<<<M, 64, 0, stream>>>(logits, a_f, a_ht);
        asum_kernel<<<B, 64, 0, stream>>>(a_f, asum, Nseq);
        // vlad[b] = aT[b] (64 x Nseq, unit k-stride) @ xn[b] (Nseq x 128)
        {
            int tiles = (64 / 16) * (128 / 16);
            dim3 grid((tiles + 7) / 8, 1, B);
            gemm_wmma_gen_kernel<<<grid, 256, 0, stream>>>(
                a_ht, xn_h, vlad, 64, 128, Nseq,
                /*sAm=*/Nseq, /*sBk=*/128, /*sBn=*/1,
                /*bA=*/64 * Nseq, /*bB=*/Nseq * 128, /*bC=*/8192);
        }
        vlad_post1_kernel<<<B * 64, 128, 0, stream>>>(vlad, asum, cent);
        vlad_post2_kernel<<<B, 256, 0, stream>>>(vlad, vlad_h);
        castw(bn_w, bn_w_h, OUT * 8192);
        gemmF32(vlad_h, bn_w_h, bn_b, bnlin, B, 256, 8192);
        desc_finish_kernel<<<B, 256, 0, stream>>>(bnlin, bn_g, bn_be,
                                                  (float*)d_out + 2072 + slot * 2048);
    };
    branch(g_emb, 30, 0);
    branch(v_emb, 37, 1);
    branch(fused, 44, 2);

    // -------- final combine --------
    final_kernel<<<B, 256, 0, stream>>>((float*)d_out);
}
